// GNN_14353780704002
// MI455X (gfx1250) — matrix operand order repaired and checked by reading, not compiled, not run on hardware
//
#include <hip/hip_runtime.h>
#include <hip/hip_bf16.h>

typedef __attribute__((ext_vector_type(16))) __bf16 bf16x16;
typedef __attribute__((ext_vector_type(2)))  __bf16 bf16x2;
typedef __attribute__((ext_vector_type(8)))  float  f32x8;

constexpr int NN   = 50000;
constexpr int EE   = 100000;
constexpr int DD   = 64;
constexpr int DEH  = 128;
constexpr int DSQ  = 4096;   // D*D
constexpr int DEMB = 256;
constexpr int GG   = 2048;
constexpr float EPSV = 1e-5f;

#if __has_builtin(__builtin_amdgcn_global_load_async_to_lds_b128) && \
    __has_builtin(__builtin_amdgcn_s_wait_asynccnt)
#define GNN_ASYNC_LDS 1
typedef int v4i_vs __attribute__((vector_size(16)));              // builtin param 0 pointee
typedef __attribute__((address_space(3))) v4i_vs LDSv4i;          // builtin param 1 pointee
#else
#define GNN_ASYNC_LDS 0
#endif

// ---------------- fp32 -> bf16 convert ----------------
__global__ void gnn_cvt_bf16(const float* __restrict__ src, __bf16* __restrict__ dst, int n) {
    int i = blockIdx.x * 256 + threadIdx.x;
    if (i < n) dst[i] = (__bf16)src[i];
}

// ---------------- h = relu(edge_attr @ W1 + b1) -> bf16 [E,128] ----------------
__global__ __launch_bounds__(128)
void gnn_edge_h(const float* __restrict__ ea, const float* __restrict__ W1,
                const float* __restrict__ b1, __bf16* __restrict__ hb) {
    __shared__ float s[DD];
    const int e = blockIdx.x, t = threadIdx.x;
    if (t < DD) s[t] = ea[(size_t)e * DD + t];
    __syncthreads();
    float acc = b1[t];
#pragma unroll 8
    for (int i = 0; i < DD; ++i) acc += s[i] * W1[i * DEH + t];
    hb[(size_t)e * DEH + t] = (__bf16)fmaxf(acc, 0.f);
}

// ---------------- W_edge[E,4096] = bf16(h @ W2 + b2), WMMA bf16 ----------------
// grid (32, 64), block 256 (8 waves). Each wave owns 16 output cols, loops edge tiles.
// A tiles staged to LDS via gfx1250 async-to-LDS copies when available.
__global__ __launch_bounds__(256)
void gnn_wedge_gemm(const __bf16* __restrict__ hb, const __bf16* __restrict__ w2b,
                    const float* __restrict__ b2, __bf16* __restrict__ wedge) {
    __shared__ uint4 As4[256];                  // 16 x 128 bf16 = 4KB
    __bf16* As = (__bf16*)As4;
    const int tid = threadIdx.x, wave = tid >> 5, lane = tid & 31, hi = lane >> 4;
    const int col = blockIdx.x * 128 + wave * 16 + (lane & 15);

    // Preload B fragments for all K=128 (4 k-steps), held in registers all loop long.
    bf16x16 bfr[4];
#pragma unroll
    for (int ks = 0; ks < 4; ++ks) {
        const int kb = ks * 32 + hi * 16;
#pragma unroll
        for (int j = 0; j < 16; ++j) bfr[ks][j] = w2b[(size_t)(kb + j) * DSQ + col];
    }
    const float bias = b2[col];

    const int ntiles = EE / 16;                 // 6250
    const int per = (ntiles + gridDim.y - 1) / gridDim.y;
    const int t0 = blockIdx.y * per;
    const int t1 = min(ntiles, t0 + per);

    const int lrow = tid >> 4, lseg = tid & 15; // cooperative A load: 16B per thread
    const int arow = lane & 15;

    for (int et = t0; et < t1; ++et) {
        const __bf16* gsrc = hb + (size_t)(et * 16 + lrow) * DEH + lseg * 8;
        __syncthreads();                        // previous tile fully consumed
#if GNN_ASYNC_LDS
        __builtin_amdgcn_global_load_async_to_lds_b128(
            (v4i_vs*)gsrc, (LDSv4i*)&As4[lrow * 16 + lseg], 0, 0);
        __builtin_amdgcn_s_wait_asynccnt(0);
#else
        As4[lrow * 16 + lseg] = *(const uint4*)gsrc;
#endif
        __syncthreads();

        f32x8 acc = {};
#pragma unroll
        for (int ks = 0; ks < 4; ++ks) {
            const int kb = ks * 32 + hi * 16;
            union { bf16x16 v; uint4 u[2]; } af;
            af.u[0] = *(const uint4*)(As + arow * DEH + kb);
            af.u[1] = *(const uint4*)(As + arow * DEH + kb + 8);
            acc = __builtin_amdgcn_wmma_f32_16x16x32_bf16(false, af.v, false, bfr[ks],
                                                          (short)0, acc, false, false);
        }
#pragma unroll
        for (int r = 0; r < 8; ++r) {
            const int m = r + hi * 8;
            wedge[(size_t)(et * 16 + m) * DSQ + col] = (__bf16)(acc[r] + bias);
        }
    }
}

// ---------------- msg[e] = x[src[e]] @ W_edge[e]; scatter-add to agg[dst] ----------------
// one wave per edge; streams bf16 W_edge (dominant HBM traffic)
__global__ __launch_bounds__(256)
void gnn_msg_scatter(const float* __restrict__ xf, const int* __restrict__ ei,
                     const __bf16* __restrict__ wedge, float* __restrict__ agg) {
    __shared__ float sx[8][DD];
    const int wave = threadIdx.x >> 5, lane = threadIdx.x & 31;
    const int e = blockIdx.x * 8 + wave;
    const int src = ei[e];
    const int dst = ei[EE + e];
    const float2 xv = *(const float2*)(xf + (size_t)src * DD + lane * 2);
    sx[wave][lane * 2]     = xv.x;
    sx[wave][lane * 2 + 1] = xv.y;
    __syncthreads();
    const __bf16* wrow = wedge + (size_t)e * DSQ;
    __builtin_prefetch(wrow, 0, 1);             // global_prefetch_b8
    float a0 = 0.f, a1 = 0.f;
#pragma unroll 4
    for (int i = 0; i < DD; ++i) {
        const float xs = sx[wave][i];
        const bf16x2 w = *(const bf16x2*)(wrow + i * DD + lane * 2);
        a0 += xs * (float)w.x;
        a1 += xs * (float)w.y;
    }
    unsafeAtomicAdd(agg + (size_t)dst * DD + lane * 2,     a0);
    unsafeAtomicAdd(agg + (size_t)dst * DD + lane * 2 + 1, a1);
}

// ---------------- x_new = x @ root + agg + bias (WMMA), + BN column stats ----------------
// grid 3125, block 128 (4 waves, one 16-col tile each)
__global__ __launch_bounds__(128)
void gnn_xroot(const __bf16* __restrict__ xb, const __bf16* __restrict__ rootb,
               const float* __restrict__ agg, const float* __restrict__ cbias,
               float* __restrict__ xtmp, float* __restrict__ bnsum, float* __restrict__ bnsq) {
    __shared__ uint4 As4[128];                  // 16 x 64 bf16 = 2KB
    __bf16* As = (__bf16*)As4;
    const int tid = threadIdx.x, wave = tid >> 5, lane = tid & 31, hi = lane >> 4;
    const int rt = blockIdx.x;
    const int lrow = tid >> 3, lseg = tid & 7;
    As4[lrow * 8 + lseg] = *(const uint4*)(xb + (size_t)(rt * 16 + lrow) * DD + lseg * 8);
    __syncthreads();
    const int col = wave * 16 + (lane & 15);
    f32x8 acc = {};
#pragma unroll
    for (int ks = 0; ks < 2; ++ks) {
        union { bf16x16 v; uint4 u[2]; } af;
        const int kb = ks * 32 + hi * 16;
        af.u[0] = *(const uint4*)(As + (lane & 15) * DD + kb);
        af.u[1] = *(const uint4*)(As + (lane & 15) * DD + kb + 8);
        bf16x16 bv;
        const int kbb = ks * 32 + hi * 16;
#pragma unroll
        for (int j = 0; j < 16; ++j) bv[j] = rootb[(kbb + j) * DD + col];
        acc = __builtin_amdgcn_wmma_f32_16x16x32_bf16(false, af.v, false, bv,
                                                      (short)0, acc, false, false);
    }
    const float cb = cbias[col];
    float s = 0.f, q = 0.f;
#pragma unroll
    for (int r = 0; r < 8; ++r) {
        const int grow = rt * 16 + r + hi * 8;
        const float v = acc[r] + agg[(size_t)grow * DD + col] + cb;
        xtmp[(size_t)grow * DD + col] = v;
        s += v; q += v * v;
    }
    s += __shfl_down(s, 16);
    q += __shfl_down(q, 16);
    if (lane < 16) {
        unsafeAtomicAdd(bnsum + col, s);
        unsafeAtomicAdd(bnsq  + col, q);
    }
}

// ---------------- BN (batch stats) + ReLU; write fp32 and bf16 copies ----------------
__global__ void gnn_bn_apply(const float* __restrict__ xtmp, const float* __restrict__ bnsum,
                             const float* __restrict__ bnsq, const float* __restrict__ gam,
                             const float* __restrict__ bet, float* __restrict__ xf,
                             __bf16* __restrict__ xb) {
    const int idx = blockIdx.x * 256 + threadIdx.x;
    if (idx >= NN * DD) return;
    const int c = idx & (DD - 1);
    const float mean = bnsum[c] * (1.0f / NN);
    const float var  = bnsq[c] * (1.0f / NN) - mean * mean;
    const float rstd = rsqrtf(var + EPSV);
    float v = gam[c] * (xtmp[idx] - mean) * rstd + bet[c];
    v = fmaxf(v, 0.f);
    xf[idx] = v;
    xb[idx] = (__bf16)v;
}

// ---------------- global_add_pool ----------------
__global__ void gnn_pool(const float* __restrict__ xf, const int* __restrict__ batch,
                         float* __restrict__ pooled) {
    const int idx = blockIdx.x * 256 + threadIdx.x;
    if (idx >= NN * DD) return;
    const int n = idx >> 6, c = idx & (DD - 1);
    unsafeAtomicAdd(pooled + (size_t)batch[n] * DD + c, xf[idx]);
}

// ---------------- out = pooled @ W_out + b_out (WMMA) ----------------
// grid (128, 4), block 128
__global__ __launch_bounds__(128)
void gnn_out_gemm(const float* __restrict__ pooled, const float* __restrict__ Wout,
                  const float* __restrict__ bout, float* __restrict__ out) {
    __shared__ uint4 As4[128];                  // 16 x 64 bf16
    __bf16* As = (__bf16*)As4;
    const int tid = threadIdx.x, wave = tid >> 5, lane = tid & 31, hi = lane >> 4;
    const int rt = blockIdx.x;
    const int lrow = tid >> 3, lseg = tid & 7;
    {
        const float* p = pooled + (size_t)(rt * 16 + lrow) * DD + lseg * 8;
        const float4 f0 = *(const float4*)(p);
        const float4 f1 = *(const float4*)(p + 4);
        __bf16* d = As + lrow * DD + lseg * 8;
        d[0] = (__bf16)f0.x; d[1] = (__bf16)f0.y; d[2] = (__bf16)f0.z; d[3] = (__bf16)f0.w;
        d[4] = (__bf16)f1.x; d[5] = (__bf16)f1.y; d[6] = (__bf16)f1.z; d[7] = (__bf16)f1.w;
    }
    __syncthreads();
    const int col = blockIdx.y * 64 + wave * 16 + (lane & 15);
    f32x8 acc = {};
#pragma unroll
    for (int ks = 0; ks < 2; ++ks) {
        union { bf16x16 v; uint4 u[2]; } af;
        const int kb = ks * 32 + hi * 16;
        af.u[0] = *(const uint4*)(As + (lane & 15) * DD + kb);
        af.u[1] = *(const uint4*)(As + (lane & 15) * DD + kb + 8);
        bf16x16 bv;
        const int kbb = ks * 32 + hi * 16;
#pragma unroll
        for (int j = 0; j < 16; ++j) bv[j] = (__bf16)Wout[(kbb + j) * DEMB + col];
        acc = __builtin_amdgcn_wmma_f32_16x16x32_bf16(false, af.v, false, bv,
                                                      (short)0, acc, false, false);
    }
    const float bb = bout[col];
#pragma unroll
    for (int r = 0; r < 8; ++r) {
        const int grow = rt * 16 + r + hi * 8;
        out[(size_t)grow * DEMB + col] = acc[r] + bb;
    }
}

extern "C" void kernel_launch(void* const* d_in, const int* in_sizes, int n_in,
                              void* d_out, int out_size, void* d_ws, size_t ws_size,
                              hipStream_t stream) {
    const float* x     = (const float*)d_in[0];
    const int*   ei    = (const int*)d_in[1];
    const float* ea    = (const float*)d_in[2];
    const int*   batch = (const int*)d_in[3];
    const float* W1    = (const float*)d_in[4];
    const float* b1    = (const float*)d_in[5];
    const float* W2    = (const float*)d_in[6];
    const float* b2    = (const float*)d_in[7];
    const float* roots = (const float*)d_in[8];
    const float* convb = (const float*)d_in[9];
    const float* gam   = (const float*)d_in[10];
    const float* bet   = (const float*)d_in[11];
    const float* Wout  = (const float*)d_in[12];
    const float* bout  = (const float*)d_in[13];
    float* out = (float*)d_out;

    char* ws = (char*)d_ws;
    size_t off = 0;
    auto carve = [&](size_t bytes) -> char* {
        char* p = ws + off;
        off += (bytes + 255) & ~(size_t)255;
        return p;
    };
    __bf16* wedge = (__bf16*)carve((size_t)EE * DSQ * 2);   // 819 MB
    __bf16* hb    = (__bf16*)carve((size_t)EE * DEH * 2);   // 25.6 MB
    __bf16* w2b   = (__bf16*)carve((size_t)DEH * DSQ * 2);  // 1 MB
    __bf16* rootb = (__bf16*)carve((size_t)3 * DD * DD * 2);
    __bf16* xb    = (__bf16*)carve((size_t)NN * DD * 2);
    float*  xf    = (float*)carve((size_t)NN * DD * 4);
    float*  xtmp  = (float*)carve((size_t)NN * DD * 4);
    float*  agg   = (float*)carve((size_t)NN * DD * 4);
    float*  bnsum = (float*)carve((size_t)2 * DD * 4);
    float*  bnsq  = bnsum + DD;
    float*  pooled= (float*)carve((size_t)GG * DD * 4);

    (void)hipMemsetAsync(pooled, 0, (size_t)GG * DD * 4, stream);

    // One-time conversions
    gnn_cvt_bf16<<<(DEH * DSQ + 255) / 256, 256, 0, stream>>>(W2, w2b, DEH * DSQ);
    gnn_cvt_bf16<<<(3 * DD * DD + 255) / 256, 256, 0, stream>>>(roots, rootb, 3 * DD * DD);
    gnn_cvt_bf16<<<(NN * DD + 255) / 256, 256, 0, stream>>>(x, xb, NN * DD);

    // Shared edge MLP
    gnn_edge_h<<<EE, 128, 0, stream>>>(ea, W1, b1, hb);
    gnn_wedge_gemm<<<dim3(32, 64), 256, 0, stream>>>(hb, w2b, b2, wedge);

    for (int l = 0; l < 3; ++l) {
        (void)hipMemsetAsync(agg, 0, (size_t)NN * DD * 4, stream);
        (void)hipMemsetAsync(bnsum, 0, (size_t)2 * DD * 4, stream);
        const float* xcur = (l == 0) ? x : xf;
        gnn_msg_scatter<<<EE / 8, 256, 0, stream>>>(xcur, ei, wedge, agg);
        gnn_xroot<<<NN / 16, 128, 0, stream>>>(xb, rootb + (size_t)l * DD * DD, agg,
                                               convb + l * DD, xtmp, bnsum, bnsq);
        gnn_bn_apply<<<(NN * DD + 255) / 256, 256, 0, stream>>>(xtmp, bnsum, bnsq,
                                                                gam + l * DD, bet + l * DD,
                                                                xf, xb);
    }

    gnn_pool<<<(NN * DD + 255) / 256, 256, 0, stream>>>(xf, batch, pooled);
    gnn_out_gemm<<<dim3(GG / 16, 4), 128, 0, stream>>>(pooled, Wout, bout, out);
}
